// BiDAFAttention_18262200943099
// MI455X (gfx1250) — matrix operand order repaired and checked by reading, not compiled, not run on hardware
//
#include <hip/hip_runtime.h>
#include <hip/hip_bf16.h>
#include <cstdint>

#define B_SZ  32
#define C_LEN 1024
#define Q_LEN 128
#define H_DIM 512
#define NEG_INF (-1e30f)

typedef __attribute__((ext_vector_type(16))) __bf16 v16bf;
typedef __attribute__((ext_vector_type(8)))  __bf16 v8bf;
typedef __attribute__((ext_vector_type(8)))  float  v8f;

// ---------------------------------------------------------------------------
// prep_A: A_k[b,c,h] = bf16(context[b,c,h] * W_cq[h,k]),  s_c[b,c,k] = <row, W_c[:,k]>
// grid (C_LEN, B_SZ), block 128
// ---------------------------------------------------------------------------
__global__ __launch_bounds__(128) void prep_A(const float* __restrict__ ctx,
                                              const float* __restrict__ w,
                                              __bf16* __restrict__ Abf,
                                              float* __restrict__ s_c) {
    const int c = blockIdx.x, b = blockIdx.y, tid = threadIdx.x;
    const float* row = ctx + ((size_t)b * C_LEN + c) * H_DIM;
    float a0 = 0.f, a1 = 0.f, a2 = 0.f;
#pragma unroll
    for (int j = 0; j < 4; ++j) {
        const int h = tid + j * 128;
        const float x = row[h];
        a0 += x * w[h * 3 + 0];
        a1 += x * w[h * 3 + 1];
        a2 += x * w[h * 3 + 2];
#pragma unroll
        for (int k = 0; k < 3; ++k) {
            const float wv = w[(2 * H_DIM + h) * 3 + k];
            Abf[(((size_t)b * 3 + k) * C_LEN + c) * H_DIM + h] = (__bf16)(x * wv);
        }
    }
    __shared__ float red0[128], red1[128], red2[128];
    red0[tid] = a0; red1[tid] = a1; red2[tid] = a2;
    __syncthreads();
    for (int off = 64; off > 0; off >>= 1) {
        if (tid < off) {
            red0[tid] += red0[tid + off];
            red1[tid] += red1[tid + off];
            red2[tid] += red2[tid + off];
        }
        __syncthreads();
    }
    if (tid == 0) {
        float* sc = s_c + ((size_t)b * C_LEN + c) * 3;
        sc[0] = red0[0]; sc[1] = red1[0]; sc[2] = red2[0];
    }
}

// ---------------------------------------------------------------------------
// prep_Q: Qbf[b,q,h] = bf16(q), Qt[b,h,q] = bf16(q) (transposed for AV GEMM),
//         s_q[b,q,k] = <row, W_q[:,k]>
// grid (Q_LEN, B_SZ), block 128
// ---------------------------------------------------------------------------
__global__ __launch_bounds__(128) void prep_Q(const float* __restrict__ que,
                                              const float* __restrict__ w,
                                              __bf16* __restrict__ Qbf,
                                              __bf16* __restrict__ Qt,
                                              float* __restrict__ s_q) {
    const int q = blockIdx.x, b = blockIdx.y, tid = threadIdx.x;
    const float* row = que + ((size_t)b * Q_LEN + q) * H_DIM;
    float a0 = 0.f, a1 = 0.f, a2 = 0.f;
#pragma unroll
    for (int j = 0; j < 4; ++j) {
        const int h = tid + j * 128;
        const float x = row[h];
        a0 += x * w[(H_DIM + h) * 3 + 0];
        a1 += x * w[(H_DIM + h) * 3 + 1];
        a2 += x * w[(H_DIM + h) * 3 + 2];
        const __bf16 xb = (__bf16)x;
        Qbf[((size_t)b * Q_LEN + q) * H_DIM + h] = xb;
        Qt[((size_t)b * H_DIM + h) * Q_LEN + q] = xb;
    }
    __shared__ float red0[128], red1[128], red2[128];
    red0[tid] = a0; red1[tid] = a1; red2[tid] = a2;
    __syncthreads();
    for (int off = 64; off > 0; off >>= 1) {
        if (tid < off) {
            red0[tid] += red0[tid + off];
            red1[tid] += red1[tid + off];
            red2[tid] += red2[tid + off];
        }
        __syncthreads();
    }
    if (tid == 0) {
        float* sq = s_q + ((size_t)b * Q_LEN + q) * 3;
        sq[0] = red0[0]; sq[1] = red1[0]; sq[2] = red2[0];
    }
}

// ---------------------------------------------------------------------------
// sim_kernel: sim[b,c,q] = sum_k tanh(s_c + s_q + (A_k @ Q^T)) + mask
// grid (C_LEN/128, B_SZ), block 256 (8 wave32s); wave = 32(M) x 64(N) of
// v_wmma_f32_16x16x32_bf16 subtiles; outer loop over k.
// ---------------------------------------------------------------------------
__global__ __launch_bounds__(256) void sim_kernel(const __bf16* __restrict__ Abf,
                                                  const __bf16* __restrict__ Qbf,
                                                  const float* __restrict__ s_c,
                                                  const float* __restrict__ s_q,
                                                  const uint8_t* __restrict__ cmask,
                                                  const uint8_t* __restrict__ qmask,
                                                  float* __restrict__ sim) {
    const int b   = blockIdx.y;
    const int c0  = blockIdx.x * 128;
    const int tid = threadIdx.x;
    const int wave = tid >> 5, lane = tid & 31;
    const int wm = wave & 3;        // M-block of 32 rows
    const int wn = wave >> 2;       // N-block of 64 cols
    const int l16 = lane & 15, hi = lane >> 4;

    v8f simsum[2][4];
    const v8f vzero = {0.f, 0.f, 0.f, 0.f, 0.f, 0.f, 0.f, 0.f};
#pragma unroll
    for (int i = 0; i < 2; ++i)
#pragma unroll
        for (int j = 0; j < 4; ++j) simsum[i][j] = vzero;

    const __bf16* Qb = Qbf + (size_t)b * Q_LEN * H_DIM;

    for (int k = 0; k < 3; ++k) {
        const __bf16* Ab = Abf + ((size_t)b * 3 + k) * C_LEN * H_DIM;
        v8f acc[2][4];
#pragma unroll
        for (int i = 0; i < 2; ++i)
#pragma unroll
            for (int j = 0; j < 4; ++j) acc[i][j] = vzero;

        for (int kk = 0; kk < H_DIM; kk += 32) {
            v16bf afrag[2];
#pragma unroll
            for (int i = 0; i < 2; ++i) {
                const __bf16* ap = Ab + ((size_t)(c0 + wm * 32 + i * 16 + l16)) * H_DIM
                                      + kk + hi * 8;
                v8bf alo = *(const v8bf*)(ap);
                v8bf ahi = *(const v8bf*)(ap + 16);
                afrag[i] = __builtin_shufflevector(alo, ahi,
                          0, 1, 2, 3, 4, 5, 6, 7, 8, 9, 10, 11, 12, 13, 14, 15);
                if (kk + 32 < H_DIM) __builtin_prefetch(ap + 32, 0, 1);
            }
            v16bf bfrag[4];
#pragma unroll
            for (int j = 0; j < 4; ++j) {
                const __bf16* bp = Qb + ((size_t)(wn * 64 + j * 16 + l16)) * H_DIM
                                      + kk + hi * 8;
                v8bf blo = *(const v8bf*)(bp);
                v8bf bhi = *(const v8bf*)(bp + 16);
                bfrag[j] = __builtin_shufflevector(blo, bhi,
                          0, 1, 2, 3, 4, 5, 6, 7, 8, 9, 10, 11, 12, 13, 14, 15);
            }
#pragma unroll
            for (int i = 0; i < 2; ++i)
#pragma unroll
                for (int j = 0; j < 4; ++j)
                    acc[i][j] = __builtin_amdgcn_wmma_f32_16x16x32_bf16(
                        false, afrag[i], false, bfrag[j],
                        (short)0, acc[i][j], false, false);
        }

        float scv[2][8];
#pragma unroll
        for (int i = 0; i < 2; ++i)
#pragma unroll
            for (int e = 0; e < 8; ++e) {
                const int c = c0 + wm * 32 + i * 16 + e + 8 * hi;
                scv[i][e] = s_c[((size_t)b * C_LEN + c) * 3 + k];
            }
        float sqv[4];
#pragma unroll
        for (int j = 0; j < 4; ++j) {
            const int q = wn * 64 + j * 16 + l16;
            sqv[j] = s_q[((size_t)b * Q_LEN + q) * 3 + k];
        }
#pragma unroll
        for (int i = 0; i < 2; ++i)
#pragma unroll
            for (int j = 0; j < 4; ++j)
#pragma unroll
                for (int e = 0; e < 8; ++e)
                    simsum[i][j][e] += tanhf(acc[i][j][e] + scv[i][e] + sqv[j]);
    }

#pragma unroll
    for (int i = 0; i < 2; ++i)
#pragma unroll
        for (int j = 0; j < 4; ++j)
#pragma unroll
            for (int e = 0; e < 8; ++e) {
                const int c = c0 + wm * 32 + i * 16 + e + 8 * hi;
                const int q = wn * 64 + j * 16 + l16;
                const float m =
                    (cmask[(size_t)b * C_LEN + c] && qmask[(size_t)b * Q_LEN + q])
                        ? 0.f : NEG_INF;
                sim[((size_t)b * C_LEN + c) * Q_LEN + q] = simsum[i][j][e] + m;
            }
}

// ---------------------------------------------------------------------------
// c2q_kernel: m[q] = max_c sim[b,c,q]; attn[b,q] = softmax_q(m)
// grid B_SZ, block 128
// ---------------------------------------------------------------------------
__global__ __launch_bounds__(128) void c2q_kernel(const float* __restrict__ sim,
                                                  float* __restrict__ attn) {
    const int b = blockIdx.x, q = threadIdx.x;
    float m = NEG_INF;
    for (int c = 0; c < C_LEN; ++c)
        m = fmaxf(m, sim[((size_t)b * C_LEN + c) * Q_LEN + q]);
    __shared__ float red[128];
    red[q] = m;
    __syncthreads();
    for (int off = 64; off > 0; off >>= 1) {
        if (q < off) red[q] = fmaxf(red[q], red[q + off]);
        __syncthreads();
    }
    const float mx = red[0];
    __syncthreads();
    const float e = __expf(m - mx);
    red[q] = e;
    __syncthreads();
    for (int off = 64; off > 0; off >>= 1) {
        if (q < off) red[q] += red[q + off];
        __syncthreads();
    }
    attn[(size_t)b * Q_LEN + q] = e / red[0];
}

// ---------------------------------------------------------------------------
// qsum_kernel: q_sum[b,h] = sum_q attn[b,q] * question[b,q,h]
// grid B_SZ, block 256
// ---------------------------------------------------------------------------
__global__ __launch_bounds__(256) void qsum_kernel(const float* __restrict__ attn,
                                                   const float* __restrict__ que,
                                                   float* __restrict__ qsum) {
    const int b = blockIdx.x;
    for (int h = threadIdx.x; h < H_DIM; h += 256) {
        float s = 0.f;
#pragma unroll 4
        for (int q = 0; q < Q_LEN; ++q)
            s += attn[(size_t)b * Q_LEN + q] * que[((size_t)b * Q_LEN + q) * H_DIM + h];
        qsum[(size_t)b * H_DIM + h] = s;
    }
}

// ---------------------------------------------------------------------------
// softmax_p_kernel: Pbf[b,c,q] = bf16(softmax_q(sim[b,c,:]))
// grid (C_LEN, B_SZ), block 128
// ---------------------------------------------------------------------------
__global__ __launch_bounds__(128) void softmax_p_kernel(const float* __restrict__ sim,
                                                        __bf16* __restrict__ Pbf) {
    const int c = blockIdx.x, b = blockIdx.y, tid = threadIdx.x;
    __shared__ float red[128];
    const float v = sim[((size_t)b * C_LEN + c) * Q_LEN + tid];
    red[tid] = v;
    __syncthreads();
    for (int off = 64; off > 0; off >>= 1) {
        if (tid < off) red[tid] = fmaxf(red[tid], red[tid + off]);
        __syncthreads();
    }
    const float mx = red[0];
    __syncthreads();
    const float e = __expf(v - mx);
    red[tid] = e;
    __syncthreads();
    for (int off = 64; off > 0; off >>= 1) {
        if (tid < off) red[tid] += red[tid + off];
        __syncthreads();
    }
    Pbf[((size_t)b * C_LEN + c) * Q_LEN + tid] = (__bf16)(e / red[0]);
}

// ---------------------------------------------------------------------------
// av_out_kernel: q2c = P @ question via WMMA (K = Q_LEN = 128), then fused
// epilogue writing all 6 G slabs for a 128(c) x 128(h) tile.
// grid (C_LEN/128, H_DIM/128, B_SZ), block 256
// ---------------------------------------------------------------------------
__global__ __launch_bounds__(256) void av_out_kernel(const __bf16* __restrict__ Pbf,
                                                     const __bf16* __restrict__ Qt,
                                                     const float* __restrict__ ctx,
                                                     const float* __restrict__ qsum,
                                                     float* __restrict__ G) {
    const int b   = blockIdx.z;
    const int c0  = blockIdx.x * 128;
    const int h0  = blockIdx.y * 128;
    const int tid = threadIdx.x;
    const int wave = tid >> 5, lane = tid & 31;
    const int wm = wave & 3;        // 32 c-rows
    const int wn = wave >> 2;       // 64 h-cols
    const int l16 = lane & 15, hi = lane >> 4;

    const v8f vzero = {0.f, 0.f, 0.f, 0.f, 0.f, 0.f, 0.f, 0.f};
    v8f acc[2][4];
#pragma unroll
    for (int i = 0; i < 2; ++i)
#pragma unroll
        for (int j = 0; j < 4; ++j) acc[i][j] = vzero;

    const __bf16* Pb  = Pbf + (size_t)b * C_LEN * Q_LEN;
    const __bf16* Qtb = Qt + (size_t)b * H_DIM * Q_LEN;

#pragma unroll
    for (int qk = 0; qk < Q_LEN; qk += 32) {
        v16bf afrag[2];
#pragma unroll
        for (int i = 0; i < 2; ++i) {
            const __bf16* ap = Pb + ((size_t)(c0 + wm * 32 + i * 16 + l16)) * Q_LEN
                                  + qk + hi * 8;
            v8bf alo = *(const v8bf*)(ap);
            v8bf ahi = *(const v8bf*)(ap + 16);
            afrag[i] = __builtin_shufflevector(alo, ahi,
                      0, 1, 2, 3, 4, 5, 6, 7, 8, 9, 10, 11, 12, 13, 14, 15);
        }
        v16bf bfrag[4];
#pragma unroll
        for (int j = 0; j < 4; ++j) {
            const __bf16* bp = Qtb + ((size_t)(h0 + wn * 64 + j * 16 + l16)) * Q_LEN
                                   + qk + hi * 8;
            v8bf blo = *(const v8bf*)(bp);
            v8bf bhi = *(const v8bf*)(bp + 16);
            bfrag[j] = __builtin_shufflevector(blo, bhi,
                      0, 1, 2, 3, 4, 5, 6, 7, 8, 9, 10, 11, 12, 13, 14, 15);
        }
#pragma unroll
        for (int i = 0; i < 2; ++i)
#pragma unroll
            for (int j = 0; j < 4; ++j)
                acc[i][j] = __builtin_amdgcn_wmma_f32_16x16x32_bf16(
                    false, afrag[i], false, bfrag[j],
                    (short)0, acc[i][j], false, false);
    }

    // fused epilogue: G[b,c, slab*H + h]
    float qsv[4];
#pragma unroll
    for (int j = 0; j < 4; ++j)
        qsv[j] = qsum[(size_t)b * H_DIM + h0 + wn * 64 + j * 16 + l16];

#pragma unroll
    for (int i = 0; i < 2; ++i)
#pragma unroll
        for (int e = 0; e < 8; ++e) {
            const int c = c0 + wm * 32 + i * 16 + e + 8 * hi;
            const size_t gbase = ((size_t)b * C_LEN + c) * (size_t)(6 * H_DIM);
            const float* crow = ctx + ((size_t)b * C_LEN + c) * H_DIM;
#pragma unroll
            for (int j = 0; j < 4; ++j) {
                const int h = h0 + wn * 64 + j * 16 + l16;
                const float q2c = acc[i][j][e];
                const float ce  = crow[h];
                const float qs  = qsv[j];
                G[gbase + 0 * H_DIM + h] = ce;
                G[gbase + 1 * H_DIM + h] = q2c;
                G[gbase + 2 * H_DIM + h] = ce * q2c;
                G[gbase + 3 * H_DIM + h] = ce * qs;
                G[gbase + 4 * H_DIM + h] = fabsf(ce - q2c);
                G[gbase + 5 * H_DIM + h] = fabsf(ce - qs);
            }
        }
}

// ---------------------------------------------------------------------------
extern "C" void kernel_launch(void* const* d_in, const int* in_sizes, int n_in,
                              void* d_out, int out_size, void* d_ws, size_t ws_size,
                              hipStream_t stream) {
    const float*   ctx   = (const float*)d_in[0];
    const float*   que   = (const float*)d_in[1];
    const uint8_t* cmask = (const uint8_t*)d_in[2];
    const uint8_t* qmask = (const uint8_t*)d_in[3];
    const float*   w     = (const float*)d_in[4];
    float* G = (float*)d_out;
    char*  ws = (char*)d_ws;

    size_t off = 0;
    __bf16* Abf = (__bf16*)(ws + off); off += (size_t)3 * B_SZ * C_LEN * H_DIM * sizeof(__bf16); // 96 MB
    __bf16* Qbf = (__bf16*)(ws + off); off += (size_t)B_SZ * Q_LEN * H_DIM * sizeof(__bf16);     // 4 MB
    __bf16* Qt  = (__bf16*)(ws + off); off += (size_t)B_SZ * H_DIM * Q_LEN * sizeof(__bf16);     // 4 MB
    __bf16* Pbf = (__bf16*)(ws + off); off += (size_t)B_SZ * C_LEN * Q_LEN * sizeof(__bf16);     // 8 MB
    float*  s_c = (float*)(ws + off);  off += (size_t)B_SZ * C_LEN * 3 * sizeof(float);
    float*  s_q = (float*)(ws + off);  off += (size_t)B_SZ * Q_LEN * 3 * sizeof(float);
    float*  sim = (float*)(ws + off);  off += (size_t)B_SZ * C_LEN * Q_LEN * sizeof(float);      // 16 MB
    float*  attn = (float*)(ws + off); off += (size_t)B_SZ * Q_LEN * sizeof(float);
    float*  qsum = (float*)(ws + off); off += (size_t)B_SZ * H_DIM * sizeof(float);
    (void)ws_size; (void)in_sizes; (void)n_in; (void)out_size;

    prep_A<<<dim3(C_LEN, B_SZ), 128, 0, stream>>>(ctx, w, Abf, s_c);
    prep_Q<<<dim3(Q_LEN, B_SZ), 128, 0, stream>>>(que, w, Qbf, Qt, s_q);
    sim_kernel<<<dim3(C_LEN / 128, B_SZ), 256, 0, stream>>>(Abf, Qbf, s_c, s_q,
                                                            cmask, qmask, sim);
    c2q_kernel<<<B_SZ, 128, 0, stream>>>(sim, attn);
    qsum_kernel<<<B_SZ, 256, 0, stream>>>(attn, que, qsum);
    softmax_p_kernel<<<dim3(C_LEN, B_SZ), 128, 0, stream>>>(sim, Pbf);
    av_out_kernel<<<dim3(C_LEN / 128, H_DIM / 128, B_SZ), 256, 0, stream>>>(
        Pbf, Qt, ctx, qsum, G);
}
